// build_model_18820546691174
// MI455X (gfx1250) — matrix operand order repaired and checked by reading, not compiled, not run on hardware
//
#include <hip/hip_runtime.h>
#include <hip/hip_bf16.h>

// ---------------------------------------------------------------------------
// MI455X (gfx1250) GRU rollout.
//   B=32, H=48, T=16, F=256, 3F=768, L=2, S=64, A=16, R=8
// Pipeline:
//   K1 pack_w      : f32 weights -> bf16 WMMA-B-fragment packed layout (x4 mats)
//   K2 embed_hist  : tanh(hist_s@Ws^T + hist_a@Wa^T + bs + ba) -> seqA (bf16)
//   K2b embed_feat : present_feat / fa_feat (f32)
//   K3 gemm_xg     : wide WMMA GEMM xg = seq @ Wih_l^T + bih_l (M=1536,N=768,K=256)
//                    4 mt-tiles per wave-job sharing B fragments (4x less B traffic)
//   K4 gru_scan    : persistent 1-WG scan, 48 steps. Column-jobs (2 mt per nt,
//                    shared B) + 192KB of Whh cached in LDS (kt 0..3) loaded once
//                    -> streamed weights drop from 768KB/step to 192KB/step.
//                    LDS: 16KB h + 96KB hg + 192KB Bcache = 304KB (<320KB WGP).
//   K5 rollout     : persistent 1-WG, 16 steps, 4 WMMA GEMMs/step (column-jobs)
//                    + heads.  LDS 248KB.
// ---------------------------------------------------------------------------

typedef __attribute__((ext_vector_type(16))) __bf16 v16bf;
typedef __attribute__((ext_vector_type(8)))  __bf16 v8bf;
typedef __attribute__((ext_vector_type(8)))  float  v8f;

#define F_DIM  256
#define G_DIM  768
#define B_SZ   32
#define H_SEQ  48
#define T_ROLL 16
#define NT_N   48   // 768/16 tiles along N
#define NT_K   8    // 256/32 tiles along K

__device__ __forceinline__ __bf16 f2bf(float f) {
  union { float f; unsigned u; } a; a.f = f;
  unsigned r = a.u + 0x7FFFu + ((a.u >> 16) & 1u);   // round-to-nearest-even
  union { unsigned short u; __bf16 b; } o; o.u = (unsigned short)(r >> 16);
  return o.b;
}
__device__ __forceinline__ float bf2f(__bf16 b) {
  union { unsigned short u; __bf16 b; } i; i.b = b;
  union { unsigned u; float f; } o; o.u = ((unsigned)i.u) << 16;
  return o.f;
}
__device__ __forceinline__ float sigf(float x) { return 1.0f / (1.0f + __expf(-x)); }
__device__ __forceinline__ float tanh_fast(float x) {
  x = fminf(15.0f, fmaxf(-15.0f, x));
  float e = __expf(2.0f * x);
  return (e - 1.0f) / (e + 1.0f);
}

// --- WMMA fragment helpers (layouts per CDNA5 ISA 7.12.2) --------
// A (16x32 bf16): lane m = lane&15; lanes<16 hold K {k0..k0+7, k0+16..k0+23},
//                 lanes>=16 hold K {k0+8..k0+15, k0+24..k0+31}.
__device__ __forceinline__ v16bf load_A(const __bf16* X, int ld, int m0, int k0, int lane) {
  int m  = m0 + (lane & 15);
  int kb = k0 + ((lane & 16) ? 8 : 0);
  const __bf16* p = X + (size_t)m * ld + kb;
  v8bf lo = *(const v8bf*)p;            // 16B contiguous
  v8bf hi = *(const v8bf*)(p + 16);     // 16B contiguous
  v16bf a;
#pragma unroll
  for (int i = 0; i < 8; ++i) { a[i] = lo[i]; a[i + 8] = hi[i]; }
  return a;
}
// B (32x16 bf16): pre-packed so each lane reads 16 contiguous halves (32B).
__device__ __forceinline__ v16bf load_B(const __bf16* Wp, int nt, int kt, int lane) {
  return *(const v16bf*)(Wp + ((((size_t)nt * NT_K + kt) * 32 + lane) << 4));
}
__device__ __forceinline__ v8f wmma_bf16(v16bf a, v16bf b, v8f c) {
  return __builtin_amdgcn_wmma_f32_16x16x32_bf16(false, a, false, b, (short)0, c, false, false);
}

// One nt column (16 N), both mt tiles of a 32-row A, sharing each B fragment.
// First KTC kt-planes of B come from an LDS cache, the rest stream from global.
template <int KTC>
__device__ __forceinline__ void gemm_col2(const __bf16* A, const __bf16* Wp,
                                          const __bf16* Bc, const float* bias,
                                          float* Y, int nt, int lane) {
  v8f c0 = {}, c1 = {};
#pragma unroll
  for (int kt = 0; kt < NT_K; ++kt) {
    v16bf b;
    if (kt < KTC)
      b = *(const v16bf*)(Bc + (((size_t)(nt * KTC + kt) * 32 + lane) << 4));
    else
      b = load_B(Wp, nt, kt, lane);
    v16bf a0 = load_A(A, F_DIM, 0,  kt * 32, lane);
    v16bf a1 = load_A(A, F_DIM, 16, kt * 32, lane);
    c0 = wmma_bf16(a0, b, c0);
    c1 = wmma_bf16(a1, b, c1);
  }
  int n  = nt * 16 + (lane & 15);
  float bv = bias[n];
  int mb = (lane & 16) ? 8 : 0;
#pragma unroll
  for (int j = 0; j < 8; ++j) {
    Y[(size_t)(mb + j) * G_DIM + n]      = c0[j] + bv;
    Y[(size_t)(16 + mb + j) * G_DIM + n] = c1[j] + bv;
  }
}

// --- K1: pack one [768,256] f32 matrix into bf16 B-fragment layout ----------
__global__ __launch_bounds__(256) void pack_w(const float* __restrict__ W,
                                              __bf16* __restrict__ P) {
  int idx = blockIdx.x * 256 + threadIdx.x;          // 196608 total
  if (idx >= NT_N * NT_K * 512) return;
  int i    = idx & 15;
  int lane = (idx >> 4) & 31;
  int kt   = (idx >> 9) & 7;
  int nt   = idx >> 12;
  int n = nt * 16 + (lane & 15);
  int k = kt * 32 + ((lane & 16) ? 8 : 0) + i + ((i & 8) ? 8 : 0);
  P[idx] = f2bf(W[(size_t)n * F_DIM + k]);
}

// --- K2: history embedding -> seqA (bf16, rows = b*48+t) --------------------
__global__ __launch_bounds__(256) void embed_hist(const float* __restrict__ hs,
                                                  const float* __restrict__ ha,
                                                  const float* __restrict__ Ws,
                                                  const float* __restrict__ bs,
                                                  const float* __restrict__ Wa,
                                                  const float* __restrict__ ba,
                                                  __bf16* __restrict__ seqA) {
  __shared__ float xs[64];
  __shared__ float xa[16];
  int row = blockIdx.x;              // b*48+t, 1536 rows
  int f   = threadIdx.x;             // 0..255
  if (f < 64) xs[f] = hs[(size_t)row * 64 + f];
  if (f < 16) xa[f] = ha[(size_t)row * 16 + f];
  __syncthreads();
  float acc = bs[f] + ba[f];
#pragma unroll 8
  for (int k = 0; k < 64; ++k) acc += xs[k] * Ws[f * 64 + k];
#pragma unroll
  for (int k = 0; k < 16; ++k) acc += xa[k] * Wa[f * 16 + k];
  seqA[(size_t)row * F_DIM + f] = f2bf(tanh_fast(acc));
}

// --- K2b: generic small affine Y[row,f] = X[row,:K]@W[f,:K]^T + b[f] --------
__global__ __launch_bounds__(256) void embed_feat(const float* __restrict__ X,
                                                  const float* __restrict__ W,
                                                  const float* __restrict__ bias,
                                                  float* __restrict__ Y, int K) {
  __shared__ float xb[64];
  int row = blockIdx.x;
  int f   = threadIdx.x;
  if (f < K) xb[f] = X[(size_t)row * K + f];
  __syncthreads();
  float acc = bias[f];
  for (int k = 0; k < K; ++k) acc += xb[k] * W[f * K + k];
  Y[(size_t)row * F_DIM + f] = acc;
}

// --- K3: wide WMMA GEMM  Y[M,768] = X[M,256] @ Wpacked + bias ---------------
// Each wave-job: one nt column x 4 mt tiles (B fragment shared 4 ways).
__global__ __launch_bounds__(256) void gemm_xg_kernel(const __bf16* __restrict__ X,
                                                      const __bf16* __restrict__ Wp,
                                                      const float* __restrict__ bias,
                                                      float* __restrict__ Y, int M) {
  int gw   = (blockIdx.x * 256 + threadIdx.x) >> 5;   // global wave id
  int lane = threadIdx.x & 31;
  int jobs = (M >> 6) * NT_N;                         // 4 mt per job
  if (gw >= jobs) return;                             // wave-uniform
  int mtg = gw / NT_N;
  int nt  = gw % NT_N;
  v8f z = {};
  v8f c[4] = { z, z, z, z };
#pragma unroll
  for (int kt = 0; kt < NT_K; ++kt) {
    v16bf b = load_B(Wp, nt, kt, lane);
#pragma unroll
    for (int m = 0; m < 4; ++m) {
      v16bf a = load_A(X, F_DIM, mtg * 64 + m * 16, kt * 32, lane);
      c[m] = wmma_bf16(a, b, c[m]);
    }
  }
  int n  = nt * 16 + (lane & 15);
  float bv = bias[n];
  int mb = (lane & 16) ? 8 : 0;
#pragma unroll
  for (int m = 0; m < 4; ++m)
#pragma unroll
    for (int j = 0; j < 8; ++j)
      Y[(size_t)(mtg * 64 + m * 16 + mb + j) * G_DIM + n] = c[m][j] + bv;
}

// --- K4: persistent 1-WG GRU scan over H=48 (one layer) ---------------------
// dyn LDS: h_bf 16KB + hg 96KB + Bcache(kt0..3 of Whh) 192KB = 304KB
__global__ __launch_bounds__(1024) void gru_scan(const float* __restrict__ xg,
                                                 const __bf16* __restrict__ WhhP,
                                                 const float* __restrict__ bhh,
                                                 __bf16* __restrict__ seq_out,
                                                 float* __restrict__ h_final) {
  extern __shared__ char smem[];
  __bf16* h_bf = (__bf16*)smem;               // [32][256]
  float*  hg   = (float*)(smem + 16384);      // [32][768]
  __bf16* Bc   = (__bf16*)(smem + 114688);    // 48nt x 4kt x 512 halves
  int tid  = threadIdx.x;
  int lane = tid & 31;
  int wave = tid >> 5;                        // 0..31
  int ob   = tid >> 5;                        // owned batch row 0..31
  int of0  = (tid & 31) * 8;                  // owned f-range [of0, of0+8)

  // one-time: cache kt-planes 0..3 of packed Whh into LDS (192KB, 32B units)
  for (int u = tid; u < NT_N * 4 * 32; u += 1024) {
    int nt = u >> 7;
    int r  = u & 127;
    int kt = r >> 5;
    int ln = r & 31;
    *(v16bf*)(Bc + ((size_t)u << 4)) =
        *(const v16bf*)(WhhP + ((((size_t)nt * NT_K + kt) * 32 + ln) << 4));
  }

  float h[8];
#pragma unroll
  for (int j = 0; j < 8; ++j) { h[j] = 0.0f; h_bf[ob * F_DIM + of0 + j] = f2bf(0.0f); }
  __syncthreads();

  for (int t = 0; t < H_SEQ; ++t) {
    // hg = h @ Whh^T + bhh : 48 column-jobs (2 mt tiles each, shared B)
    for (int nt = wave; nt < NT_N; nt += 32)
      gemm_col2<4>(h_bf, WhhP, Bc, bhh, hg, nt, lane);
    __syncthreads();
    // gates (r,z,n) for owned elements
    const float* xrow = xg + (size_t)(ob * H_SEQ + t) * G_DIM;
    const float* hrow = hg + (size_t)ob * G_DIM;
#pragma unroll
    for (int j = 0; j < 8; ++j) {
      int f = of0 + j;
      float r  = sigf(xrow[f]           + hrow[f]);
      float z  = sigf(xrow[F_DIM + f]   + hrow[F_DIM + f]);
      float nn = tanh_fast(xrow[2*F_DIM + f] + r * hrow[2*F_DIM + f]);
      h[j] = (1.0f - z) * nn + z * h[j];
      __bf16 hb = f2bf(h[j]);
      h_bf[ob * F_DIM + f] = hb;
      seq_out[(size_t)(ob * H_SEQ + t) * F_DIM + f] = hb;
    }
    __syncthreads();
  }
#pragma unroll
  for (int j = 0; j < 8; ++j) h_final[ob * F_DIM + of0 + j] = h[j];
}

// --- K5: persistent 1-WG rollout, 16 steps ----------------------------------
// dyn LDS: x_bf/h0_bf/h1_bf (3x16KB) + xgb/hgb (2x96KB) + sbuf (8KB) = 248KB
__global__ __launch_bounds__(1024) void rollout(const __bf16* __restrict__ WihP0,
                                                const __bf16* __restrict__ WhhP0,
                                                const __bf16* __restrict__ WihP1,
                                                const __bf16* __restrict__ WhhP1,
                                                const float* __restrict__ bih,   // [2][768]
                                                const float* __restrict__ bhh,   // [2][768]
                                                const float* __restrict__ fa,    // [B*T,256]
                                                const float* __restrict__ pfeat, // [B,256]
                                                const float* __restrict__ hfin,  // [2][B,256]
                                                const float* __restrict__ Wr,
                                                const float* __restrict__ br,
                                                const float* __restrict__ Wso,
                                                const float* __restrict__ bso,
                                                const float* __restrict__ Ws,
                                                const float* __restrict__ bs,
                                                float* __restrict__ out) {
  extern __shared__ char smem[];
  __bf16* x_bf  = (__bf16*)smem;                       // [32][256]
  __bf16* h0_bf = (__bf16*)(smem + 16384);
  __bf16* h1_bf = (__bf16*)(smem + 32768);
  float*  xgb   = (float*)(smem + 49152);              // [32][768]
  float*  hgb   = (float*)(smem + 49152 + 98304);      // [32][768]
  float*  sbuf  = (float*)(smem + 49152 + 196608);     // [32][64]
  float* out_r = out;                                  // [B,T,8]
  float* out_s = out + B_SZ * T_ROLL * 8;              // [B,T,64]

  int tid  = threadIdx.x;
  int lane = tid & 31;
  int wave = tid >> 5;
  int ob   = tid >> 5;
  int of0  = (tid & 31) * 8;

  float pf[8], h0[8], h1[8];
#pragma unroll
  for (int j = 0; j < 8; ++j) {
    int f = of0 + j;
    pf[j] = pfeat[ob * F_DIM + f];
    h0[j] = hfin[ob * F_DIM + f];
    h1[j] = hfin[B_SZ * F_DIM + ob * F_DIM + f];
    h0_bf[ob * F_DIM + f] = f2bf(h0[j]);
    h1_bf[ob * F_DIM + f] = f2bf(h1[j]);
  }
  __syncthreads();

  for (int t = 0; t < T_ROLL; ++t) {
    // x = tanh(pf + fa_t)
#pragma unroll
    for (int j = 0; j < 8; ++j) {
      int f = of0 + j;
      x_bf[ob * F_DIM + f] = f2bf(tanh_fast(pf[j] + fa[(size_t)(ob * T_ROLL + t) * F_DIM + f]));
    }
    __syncthreads();
    // layer0: xg0 = x@Wih0 (+bih0), hg0 = h0@Whh0 (+bhh0): 96 column-jobs
    for (int job = wave; job < 2 * NT_N; job += 32) {
      int which = job >= NT_N;                         // wave-uniform
      int nt    = job - which * NT_N;
      gemm_col2<0>(which ? h0_bf : x_bf,
                   which ? WhhP0 : WihP0, (const __bf16*)0,
                   which ? bhh : bih,
                   which ? hgb : xgb, nt, lane);
    }
    __syncthreads();
#pragma unroll
    for (int j = 0; j < 8; ++j) {
      int f = of0 + j;
      const float* xr = xgb + (size_t)ob * G_DIM;
      const float* hr = hgb + (size_t)ob * G_DIM;
      float r  = sigf(xr[f] + hr[f]);
      float z  = sigf(xr[F_DIM + f] + hr[F_DIM + f]);
      float nn = tanh_fast(xr[2*F_DIM + f] + r * hr[2*F_DIM + f]);
      h0[j] = (1.0f - z) * nn + z * h0[j];
      h0_bf[ob * F_DIM + f] = f2bf(h0[j]);
    }
    __syncthreads();
    // layer1: xg1 = h0@Wih1 (+bih1), hg1 = h1@Whh1 (+bhh1)
    for (int job = wave; job < 2 * NT_N; job += 32) {
      int which = job >= NT_N;
      int nt    = job - which * NT_N;
      gemm_col2<0>(which ? h1_bf : h0_bf,
                   which ? WhhP1 : WihP1, (const __bf16*)0,
                   which ? (bhh + G_DIM) : (bih + G_DIM),
                   which ? hgb : xgb, nt, lane);
    }
    __syncthreads();
#pragma unroll
    for (int j = 0; j < 8; ++j) {
      int f = of0 + j;
      const float* xr = xgb + (size_t)ob * G_DIM;
      const float* hr = hgb + (size_t)ob * G_DIM;
      float r  = sigf(xr[f] + hr[f]);
      float z  = sigf(xr[F_DIM + f] + hr[F_DIM + f]);
      float nn = tanh_fast(xr[2*F_DIM + f] + r * hr[2*F_DIM + f]);
      h1[j] = (1.0f - z) * nn + z * h1[j];
      h1_bf[ob * F_DIM + f] = f2bf(h1[j]);
    }
    __syncthreads();
    // heads: r = tanh(h1@Wr^T+br) [32,8]; s = tanh(h1@Wso^T+bso) [32,64]
    if (tid < 256) {
      int b = tid >> 3, o = tid & 7;
      float acc = br[o];
      for (int k = 0; k < F_DIM; ++k) acc += bf2f(h1_bf[b * F_DIM + k]) * Wr[o * F_DIM + k];
      out_r[(size_t)(b * T_ROLL + t) * 8 + o] = tanh_fast(acc);
    }
#pragma unroll
    for (int e = 0; e < 2; ++e) {
      int idx = tid * 2 + e;
      int b = idx >> 6, o = idx & 63;
      float acc = bso[o];
      for (int k = 0; k < F_DIM; ++k) acc += bf2f(h1_bf[b * F_DIM + k]) * Wso[o * F_DIM + k];
      float v = tanh_fast(acc);
      out_s[(size_t)(b * T_ROLL + t) * 64 + o] = v;
      sbuf[b * 64 + o] = v;
    }
    __syncthreads();
    // pf = s @ Ws^T + bs
#pragma unroll
    for (int j = 0; j < 8; ++j) {
      int f = of0 + j;
      float acc = bs[f];
      for (int q = 0; q < 64; ++q) acc += sbuf[ob * 64 + q] * Ws[f * 64 + q];
      pf[j] = acc;
    }
    __syncthreads();
  }
}

// ---------------------------------------------------------------------------
extern "C" void kernel_launch(void* const* d_in, const int* in_sizes, int n_in,
                              void* d_out, int out_size, void* d_ws, size_t ws_size,
                              hipStream_t stream) {
  (void)in_sizes; (void)n_in; (void)out_size; (void)ws_size;
  const float* history_s = (const float*)d_in[0];
  const float* history_a = (const float*)d_in[1];
  const float* present_s = (const float*)d_in[2];
  const float* future_a  = (const float*)d_in[3];
  const float* Ws  = (const float*)d_in[4];
  const float* bs  = (const float*)d_in[5];
  const float* Wa  = (const float*)d_in[6];
  const float* ba  = (const float*)d_in[7];
  const float* Wih = (const float*)d_in[8];    // [2,768,256]
  const float* Whh = (const float*)d_in[9];    // [2,768,256]
  const float* bih = (const float*)d_in[10];   // [2,768]
  const float* bhh = (const float*)d_in[11];   // [2,768]
  const float* Wr  = (const float*)d_in[12];
  const float* br  = (const float*)d_in[13];
  const float* Wso = (const float*)d_in[14];
  const float* bso = (const float*)d_in[15];
  float* out = (float*)d_out;

  // workspace layout (bytes)
  char* ws = (char*)d_ws;
  const size_t PACKB = 196608 * sizeof(__bf16);       // 393216 per matrix
  __bf16* WihP0 = (__bf16*)(ws + 0 * PACKB);
  __bf16* WihP1 = (__bf16*)(ws + 1 * PACKB);
  __bf16* WhhP0 = (__bf16*)(ws + 2 * PACKB);
  __bf16* WhhP1 = (__bf16*)(ws + 3 * PACKB);
  size_t off = 4 * PACKB;
  __bf16* seqA = (__bf16*)(ws + off); off += (size_t)1536 * 256 * sizeof(__bf16);
  __bf16* seqB = (__bf16*)(ws + off); off += (size_t)1536 * 256 * sizeof(__bf16);
  float*  xg   = (float*)(ws + off);  off += (size_t)1536 * 768 * sizeof(float);
  float*  hfin = (float*)(ws + off);  off += (size_t)2 * 32 * 256 * sizeof(float);
  float*  pft  = (float*)(ws + off);  off += (size_t)32 * 256 * sizeof(float);
  float*  faf  = (float*)(ws + off);  off += (size_t)512 * 256 * sizeof(float);

  const int MAT = 768 * 256;

  // K1: pack weights (f32 -> bf16 B-fragment layout)
  pack_w<<<768, 256, 0, stream>>>(Wih,       WihP0);
  pack_w<<<768, 256, 0, stream>>>(Wih + MAT, WihP1);
  pack_w<<<768, 256, 0, stream>>>(Whh,       WhhP0);
  pack_w<<<768, 256, 0, stream>>>(Whh + MAT, WhhP1);

  // K2: embeddings
  embed_hist<<<1536, 256, 0, stream>>>(history_s, history_a, Ws, bs, Wa, ba, seqA);
  embed_feat<<<32, 256, 0, stream>>>(present_s, Ws, bs, pft, 64);
  embed_feat<<<512, 256, 0, stream>>>(future_a, Wa, ba, faf, 16);

  // layer 0: wide xg GEMM (1152 jobs, 8 waves/block -> 144 blocks), then scan
  gemm_xg_kernel<<<144, 256, 0, stream>>>(seqA, WihP0, bih, xg, 1536);
  gru_scan<<<1, 1024, 311296, stream>>>(xg, WhhP0, bhh, seqB, hfin);

  // layer 1
  gemm_xg_kernel<<<144, 256, 0, stream>>>(seqB, WihP1, bih + 768, xg, 1536);
  gru_scan<<<1, 1024, 311296, stream>>>(xg, WhhP1, bhh + 768, seqA, hfin + 32 * 256);

  // rollout (persistent, 248KB LDS on one WGP)
  rollout<<<1, 1024, 253952, stream>>>(WihP0, WhhP0, WihP1, WhhP1, bih, bhh,
                                       faf, pft, hfin, Wr, br, Wso, bso, Ws, bs, out);
}
